// BiaffineDependencyParser_85280870629901
// MI455X (gfx1250) — matrix operand order repaired and checked by reading, not compiled, not run on hardware
//
#include <hip/hip_runtime.h>
#include <hip/hip_bf16.h>
#include <math.h>

typedef __bf16 bhalf;
typedef __attribute__((ext_vector_type(16))) __bf16 v16bf;
typedef __attribute__((ext_vector_type(8)))  float  v8f;
typedef __attribute__((ext_vector_type(4)))  unsigned int v4u;

union FragBF { v16bf v; v4u u[2]; };

// ---------------------------------------------------------------------------
// WMMA fragment loaders for v_wmma_f32_16x16x32_bf16 (wave32).
// A (16x32, MxK), 16-bit layout (ISA 7.12.2):
//   lane L: m = L%16 ; element i holds k = (i/8)*16 + (L/16)*8 + (i%8)
// B (32x16, KxN) stored as Bt row-major [N,K] (NT GEMM):
//   lane L: n = L%16 ; element i holds k = (i/8)*16 + (L/16)*8 + (i%8)
// C/D (16x16 f32): lane L: n = L%16 ; vgpr v: m = v + 8*(L/16)
// ---------------------------------------------------------------------------
__device__ __forceinline__ v16bf load_frag_a(const bhalf* A, int m0, int k0,
                                             int ldk, int lane) {
    int m  = m0 + (lane & 15);
    int kh = (lane >> 4) << 3;
    const bhalf* p = A + (long)m * ldk + k0 + kh;
    FragBF f;
    f.u[0] = *(const v4u*)(p);
    f.u[1] = *(const v4u*)(p + 16);
    return f.v;
}

__device__ __forceinline__ v16bf load_frag_b(const bhalf* Bt, int n0, int k0,
                                             int ldk, int lane) {
    int n  = n0 + (lane & 15);
    int kb = (lane >> 4) << 3;
    const bhalf* p = Bt + (long)n * ldk + k0 + kb;
    FragBF f;
    f.u[0] = *(const v4u*)(p);
    f.u[1] = *(const v4u*)(p + 16);
    return f.v;
}

// ---------------------------------------------------------------------------
// Generic batched NT GEMM: C[b] = act(A[b](MxK,row) * Bt[b](NxK,row)^T + bias)
// One wave computes a 16(M) x 64(N) tile (4 accumulators).
// ---------------------------------------------------------------------------
__global__ void gemm_nt_kernel(const bhalf* __restrict__ A, long sA,
                               const bhalf* __restrict__ Bt, long sB,
                               float* __restrict__ Cf, bhalf* __restrict__ Cb,
                               long sC, const float* __restrict__ bias,
                               int M, int N, int K, int act) {
    int lane  = threadIdx.x & 31;
    int wid   = (int)((blockIdx.x * blockDim.x + threadIdx.x) >> 5);
    int numMT = M >> 4;
    int mt    = wid % numMT;
    int nt64  = wid / numMT;
    int m0 = mt << 4, n0 = nt64 << 6;
    if (n0 >= N) return;

    const bhalf* Ab = A  + (long)blockIdx.y * sA;
    const bhalf* Bb = Bt + (long)blockIdx.y * sB;

    v8f zero = {0.f,0.f,0.f,0.f,0.f,0.f,0.f,0.f};
    v8f acc[4];
    acc[0] = zero; acc[1] = zero; acc[2] = zero; acc[3] = zero;

    for (int k0 = 0; k0 < K; k0 += 32) {
        if (k0 + 32 < K)
            __builtin_prefetch(Ab + (long)(m0 + (lane & 15)) * K + k0 + 32, 0, 1);
        v16bf afr = load_frag_a(Ab, m0, k0, K, lane);
#pragma unroll
        for (int t = 0; t < 4; ++t) {
            v16bf bfr = load_frag_b(Bb, n0 + (t << 4), k0, K, lane);
            acc[t] = __builtin_amdgcn_wmma_f32_16x16x32_bf16(
                false, afr, false, bfr, (short)0, acc[t], false, false);
        }
    }

    int mb = m0 + ((lane >> 4) << 3);
#pragma unroll
    for (int t = 0; t < 4; ++t) {
        int n = n0 + (t << 4) + (lane & 15);
        float bv = bias ? bias[n] : 0.0f;
#pragma unroll
        for (int v = 0; v < 8; ++v) {
            float x = acc[t][v] + bv;
            if (act) x = (x > 0.0f) ? x : (expf(x) - 1.0f);  // ELU
            long idx = (long)blockIdx.y * sC + (long)(mb + v) * N + n;
            if (Cb) Cb[idx] = (bhalf)x;
            else    Cf[idx] = x;
        }
    }
}

// ---------------------------------------------------------------------------
// Fused label bilinear: one workgroup per (l, b).
//   At entry: issue GLOBAL_LOAD_ASYNC_TO_LDS_B128 to stage D_lab[b] (32 KB)
//             into LDS, overlapped with stage 1 (ASYNCcnt path).
//   Stage1: G = H_lab[b] (128x128) * Ulab_t[l]^T  -> LDS (bf16)
//   Stage2: out[b,l,i,j] = G[i,:] dot DL[j,:]  with B-fragments from LDS
// ---------------------------------------------------------------------------
__global__ void lab_fused_kernel(const bhalf* __restrict__ Hlab,
                                 const bhalf* __restrict__ Dlab,
                                 const bhalf* __restrict__ UlabT,
                                 float* __restrict__ out) {
    __shared__ __align__(16) bhalf G[128 * 128];
    __shared__ __align__(16) bhalf DL[128 * 128];
    int l = blockIdx.x, b = blockIdx.y;
    int wid = threadIdx.x >> 5, lane = threadIdx.x & 31;
    const bhalf* Hb = Hlab  + (long)b * 128 * 128;
    const bhalf* Db = Dlab  + (long)b * 128 * 128;
    const bhalf* Ul = UlabT + (long)l * 128 * 128;

    // --- async bulk copy D_lab[b] -> LDS (8 x b128 per thread, 32 KB total),
    //     overlapped with stage-1 WMMA work; tracked by ASYNCcnt.
    {
        unsigned ldsbase = (unsigned)(uintptr_t)(&DL[0]);  // LDS aperture: low 32b
        int tid = threadIdx.x;
#pragma unroll
        for (int r = 0; r < 8; ++r) {
            unsigned off   = (unsigned)((tid + (r << 8)) << 4);   // 16B chunks
            unsigned laddr = ldsbase + off;
            const char* gaddr = (const char*)Db + off;
            asm volatile("global_load_async_to_lds_b128 %0, %1, off"
                         :: "v"(laddr), "v"(gaddr) : "memory");
        }
    }

    int m0 = wid << 4;
    v8f zero = {0.f,0.f,0.f,0.f,0.f,0.f,0.f,0.f};

    v16bf afr[4];
#pragma unroll
    for (int kk = 0; kk < 4; ++kk) afr[kk] = load_frag_a(Hb, m0, kk * 32, 128, lane);

    int nn = lane & 15;
    int mb = m0 + ((lane >> 4) << 3);
#pragma unroll
    for (int nt = 0; nt < 8; ++nt) {
        v8f acc = zero;
#pragma unroll
        for (int kk = 0; kk < 4; ++kk) {
            v16bf bfr = load_frag_b(Ul, nt << 4, kk * 32, 128, lane);
            acc = __builtin_amdgcn_wmma_f32_16x16x32_bf16(
                false, afr[kk], false, bfr, (short)0, acc, false, false);
        }
        int n = (nt << 4) + nn;
#pragma unroll
        for (int v = 0; v < 8; ++v) G[(mb + v) * 128 + n] = (bhalf)acc[v];
    }

    // all async LDS writes complete + cross-wave visibility
    asm volatile("s_wait_asynccnt 0x0" ::: "memory");
    __syncthreads();

    v16bf gfr[4];
#pragma unroll
    for (int kk = 0; kk < 4; ++kk) gfr[kk] = load_frag_a(G, m0, kk * 32, 128, lane);

    long obase = ((long)b * 50 + l) * 16384;
#pragma unroll
    for (int nt = 0; nt < 8; ++nt) {
        v8f acc = zero;
#pragma unroll
        for (int kk = 0; kk < 4; ++kk) {
            v16bf bfr = load_frag_b(DL, nt << 4, kk * 32, 128, lane);
            acc = __builtin_amdgcn_wmma_f32_16x16x32_bf16(
                false, gfr[kk], false, bfr, (short)0, acc, false, false);
        }
        int n = (nt << 4) + nn;
#pragma unroll
        for (int v = 0; v < 8; ++v)
            out[obase + (long)(mb + v) * 128 + n] = acc[v];
    }
}

// ---------------------------------------------------------------------------
// Conversions fp32 -> bf16 (plain and batched-transpose)
// ---------------------------------------------------------------------------
__global__ void cvt_kernel(const float* __restrict__ s, bhalf* __restrict__ d, long n) {
    long i = (long)blockIdx.x * blockDim.x + threadIdx.x;
    if (i < n) d[i] = (bhalf)s[i];
}

__global__ void cvtT_kernel(const float* __restrict__ s, bhalf* __restrict__ d,
                            int rows, int cols, long total) {
    long i = (long)blockIdx.x * blockDim.x + threadIdx.x;
    if (i >= total) return;
    long per = (long)rows * cols;
    long bidx = i / per;
    int  rem  = (int)(i - bidx * per);
    int r = rem / cols, c = rem % cols;
    d[bidx * per + (long)c * rows + r] = (bhalf)s[i];
}

// ---------------------------------------------------------------------------
// arc log_softmax over heads (axis i): one thread per (b, j)
// ---------------------------------------------------------------------------
__global__ void arc_logsoftmax_kernel(const float* __restrict__ arc,
                                      float* __restrict__ norm) {
    int t = blockIdx.x * blockDim.x + threadIdx.x;   // b*T + j
    if (t >= 32 * 128) return;
    int b = t >> 7, j = t & 127;
    const float* base = arc + (long)b * 16384 + j;
    float mx = -INFINITY;
    for (int i = 0; i < 128; ++i) mx = fmaxf(mx, base[i * 128]);
    float s = 0.f;
    for (int i = 0; i < 128; ++i) s += expf(base[i * 128] - mx);
    float lse = mx + logf(s);
    float* nb = norm + (long)b * 16384 + j;
    for (int i = 0; i < 128; ++i) nb[i * 128] = base[i * 128] - lse;
}

// ---------------------------------------------------------------------------
// Label logsumexp + combine: energy = exp(norm_arc + lab - lse_l(lab)),
// in place over d_out; also record max/argmax over labels for the decoder.
// One thread per (b,i,j).
// ---------------------------------------------------------------------------
__global__ void lab_combine_kernel(float* __restrict__ energy,
                                   const float* __restrict__ norm_arc,
                                   float* __restrict__ smax,
                                   int* __restrict__ sarg) {
    long t = (long)blockIdx.x * blockDim.x + threadIdx.x;
    const long TT = 16384;
    if (t >= 32 * TT) return;
    long b = t / TT, rem = t - b * TT;
    float* col = energy + b * 50 * TT + rem;
    float mx = -INFINITY;
    for (int l = 0; l < 50; ++l) mx = fmaxf(mx, col[l * TT]);
    float s = 0.f;
    for (int l = 0; l < 50; ++l) s += expf(col[l * TT] - mx);
    float lse = mx + logf(s);
    float na = norm_arc[t];
    float bmax = -INFINITY; int barg = 0;
    for (int l = 0; l < 50; ++l) {
        float e = expf(col[l * TT] - lse + na);
        col[l * TT] = e;
        if (e > bmax) { bmax = e; barg = l; }
    }
    smax[t] = bmax;
    sarg[t] = barg;
}

// ---------------------------------------------------------------------------
// Greedy head/label decode (per-batch block, thread = child j)
// ---------------------------------------------------------------------------
__global__ void decode_kernel(const float* __restrict__ smax,
                              const int* __restrict__ sarg,
                              float* __restrict__ heads,
                              float* __restrict__ tags) {
    int b = blockIdx.x, j = threadIdx.x;
    const float* S = smax + (long)b * 16384;
    const int*   G = sarg + (long)b * 16384;
    if (j == 0) {
        heads[b * 128] = -1.0f;                       // root: final_edges[0] = -1
        tags[b * 128]  = (float)G[127 * 128];         // label_id[-1, 0] (numpy wrap)
    } else {
        float best = S[j]; int bi = 0;
        for (int i = 1; i < 128; ++i) {
            if (i == j) continue;
            float v = S[i * 128 + j];
            if (v > best) { best = v; bi = i; }
        }
        heads[b * 128 + j] = (float)bi;
        tags[b * 128 + j]  = (float)G[bi * 128 + j];
    }
}

// ---------------------------------------------------------------------------
extern "C" void kernel_launch(void* const* d_in, const int* in_sizes, int n_in,
                              void* d_out, int out_size, void* d_ws, size_t ws_size,
                              hipStream_t stream) {
    (void)in_sizes; (void)n_in; (void)out_size; (void)ws_size;
    const float* X   = (const float*)d_in[0];
    const float* Wha = (const float*)d_in[1];
    const float* bha = (const float*)d_in[2];
    const float* Wda = (const float*)d_in[3];
    const float* bda = (const float*)d_in[4];
    const float* Uar = (const float*)d_in[5];
    const float* Whl = (const float*)d_in[6];
    const float* bhl = (const float*)d_in[7];
    const float* Wdl = (const float*)d_in[8];
    const float* bdl = (const float*)d_in[9];
    const float* Ulb = (const float*)d_in[10];

    constexpr int Bn = 32, T = 128, D = 768, A = 512, R = 128, L = 50;
    constexpr int M = Bn * T;                  // 4096

    char* w = (char*)d_ws;
    auto wsAlloc = [&](size_t bytes) -> char* {
        char* p = w; w += (bytes + 255) & ~(size_t)255; return p;
    };
    bhalf* Xb    = (bhalf*)wsAlloc((size_t)M * D * 2);
    bhalf* WhaT  = (bhalf*)wsAlloc((size_t)A * D * 2);
    bhalf* WdaT  = (bhalf*)wsAlloc((size_t)A * D * 2);
    bhalf* WhlT  = (bhalf*)wsAlloc((size_t)R * D * 2);
    bhalf* WdlT  = (bhalf*)wsAlloc((size_t)R * D * 2);
    bhalf* UarT  = (bhalf*)wsAlloc((size_t)A * A * 2);
    bhalf* UlbT  = (bhalf*)wsAlloc((size_t)L * R * R * 2);
    bhalf* Harc  = (bhalf*)wsAlloc((size_t)M * A * 2);
    bhalf* Darc  = (bhalf*)wsAlloc((size_t)M * A * 2);
    bhalf* Hlab  = (bhalf*)wsAlloc((size_t)M * R * 2);
    bhalf* Dlab  = (bhalf*)wsAlloc((size_t)M * R * 2);
    bhalf* Tarc  = (bhalf*)wsAlloc((size_t)M * A * 2);
    float* arc   = (float*)wsAlloc((size_t)Bn * T * T * 4);   // later reused: smax
    float* narc  = (float*)wsAlloc((size_t)Bn * T * T * 4);
    int*   sarg  = (int*)  wsAlloc((size_t)Bn * T * T * 4);

    float* energy = (float*)d_out;                        // (B,L,T,T)
    float* heads  = energy + (long)Bn * L * T * T;        // (B,T) as float
    float* tags   = heads  + (long)Bn * T;

    // --- 1) fp32 -> bf16 conversions (weights transposed for NT GEMMs) ---
    {
        long n = (long)M * D;
        cvt_kernel<<<dim3((unsigned)((n + 255) / 256)), dim3(256), 0, stream>>>(X, Xb, n);
    }
    auto cvtT = [&](const float* s, bhalf* d, int rows, int cols, int batch) {
        long tot = (long)batch * rows * cols;
        cvtT_kernel<<<dim3((unsigned)((tot + 255) / 256)), dim3(256), 0, stream>>>(
            s, d, rows, cols, tot);
    };
    cvtT(Wha, WhaT, D, A, 1);
    cvtT(Wda, WdaT, D, A, 1);
    cvtT(Whl, WhlT, D, R, 1);
    cvtT(Wdl, WdlT, D, R, 1);
    cvtT(Uar, UarT, A, A, 1);
    cvtT(Ulb, UlbT, R, R, L);

    // --- 2) projections + ELU, then arc bilinear chain ---
    auto gemm = [&](const bhalf* Aq, long sA, const bhalf* Bt, long sB,
                    float* Cf, bhalf* Cb, long sC, const float* bias,
                    int Mq, int Nq, int Kq, int act, int batch) {
        int waves = (Mq / 16) * (Nq / 64);
        gemm_nt_kernel<<<dim3((unsigned)(waves / 8), (unsigned)batch),
                         dim3(256), 0, stream>>>(Aq, sA, Bt, sB, Cf, Cb, sC,
                                                 bias, Mq, Nq, Kq, act);
    };
    gemm(Xb, 0, WhaT, 0, nullptr, Harc, 0, bha, M, A, D, 1, 1);   // h_arc
    gemm(Xb, 0, WdaT, 0, nullptr, Darc, 0, bda, M, A, D, 1, 1);   // d_arc
    gemm(Xb, 0, WhlT, 0, nullptr, Hlab, 0, bhl, M, R, D, 1, 1);   // h_lab
    gemm(Xb, 0, WdlT, 0, nullptr, Dlab, 0, bdl, M, R, D, 1, 1);   // d_lab
    gemm(Harc, 0, UarT, 0, nullptr, Tarc, 0, nullptr, M, A, A, 0, 1);  // h@U
    gemm(Tarc, (long)T * A, Darc, (long)T * A, arc, nullptr,
         (long)T * T, nullptr, T, T, A, 0, Bn);                   // arc scores

    // --- 3) arc log_softmax over heads ---
    arc_logsoftmax_kernel<<<dim3(16), dim3(256), 0, stream>>>(arc, narc);

    // --- 4) label bilinear (raw scores into d_out energy region) ---
    lab_fused_kernel<<<dim3(L, Bn), dim3(256), 0, stream>>>(Hlab, Dlab, UlbT, energy);

    // --- 5) label logsumexp + exp-combine in place; record max/argmax_l ---
    {
        long tot = (long)Bn * T * T;
        lab_combine_kernel<<<dim3((unsigned)((tot + 255) / 256)), dim3(256),
                             0, stream>>>(energy, narc, arc, sarg);
    }

    // --- 6) greedy decode ---
    decode_kernel<<<dim3(Bn), dim3(T), 0, stream>>>(arc, sarg, heads, tags);
}
